// SelfAttention_60782377173242
// MI455X (gfx1250) — matrix operand order repaired and checked by reading, not compiled, not run on hardware
//
#include <hip/hip_runtime.h>

// ---------------------------------------------------------------------------
// Self-attention (B=2, T=2048, D=1024, H=16, HD=64) for MI455X / gfx1250.
// All matmuls via v_wmma_f32_16x16x32_f16 (fp32 accumulate). Entire working
// set (~50MB f16) fits in 192MB L2, so fragments load straight from global;
// WMMA issue is the bottleneck. GEMM K-loop is ping-pong double-buffered
// (unrolled x2, explicit buffer roles) so next-step loads overlap the matrix
// pipe with NO rotation register copies.
// Workspace layout (needs 40 MB):
//   [0,8M)    xh (f16 x)        -- reused for ctx after V projection
//   [8M,16M)  Wq^T,Wk^T,Wv^T,Wo^T (f16, column-major, 2MB each)
//   [16M,24M) Q (f16, [B,T,D])
//   [24M,32M) K (f16, [B,T,D])
//   [32M,40M) V^T (f16, [B,H,HD,T])
// ---------------------------------------------------------------------------

typedef _Float16 v8h  __attribute__((ext_vector_type(8)));
typedef _Float16 v16h __attribute__((ext_vector_type(16)));
typedef float    v8f  __attribute__((ext_vector_type(8)));

#define B_  2
#define T_  2048
#define D_  1024
#define H_  16
#define HD_ 64

static __device__ __forceinline__ v16h cat8(v8h lo, v8h hi) {
  return __builtin_shufflevector(lo, hi, 0,1,2,3,4,5,6,7,8,9,10,11,12,13,14,15);
}
// Two 16B loads -> one 16-element f16 fragment (elements [0..7] = first b128).
static __device__ __forceinline__ v16h ld2x128(const _Float16* p0, const _Float16* p1) {
  return cat8(*(const v8h*)p0, *(const v8h*)p1);
}

// ---------------------------------------------------------------------------
// f32 -> f16 convert
// ---------------------------------------------------------------------------
__global__ void cvt_f32_f16(const float* __restrict__ in, _Float16* __restrict__ out, int n) {
  int i = blockIdx.x * blockDim.x + threadIdx.x;
  if (i < n) out[i] = (_Float16)in[i];
}

// W (DxD row-major f32) -> W^T (column-major f16): Wt[n*D + k] = W[k*D + n]
__global__ void cvtT_f32_f16(const float* __restrict__ in, _Float16* __restrict__ out) {
  int i = blockIdx.x * blockDim.x + threadIdx.x;     // i < D*D
  int k = i >> 10, n = i & (D_ - 1);
  out[((size_t)n << 10) + k] = (_Float16)in[i];
}

// ---------------------------------------------------------------------------
// RoPE, in place on f16 [B,T,D]; one thread per (b,t,h, j<32) rotation pair.
// ---------------------------------------------------------------------------
__global__ void rope_f16(_Float16* __restrict__ buf,
                         const float* __restrict__ cosb,
                         const float* __restrict__ sinb) {
  int i = blockIdx.x * blockDim.x + threadIdx.x;     // B*T*H*32 threads
  int j = i & 31;
  int h = (i >> 5) & (H_ - 1);
  int t = (i >> 9) & (T_ - 1);
  int b = i >> 20;
  size_t base = (((size_t)(b * T_ + t)) << 10) + h * HD_ + j;
  float c1 = cosb[t * HD_ + j],      s1 = sinb[t * HD_ + j];
  float c2 = cosb[t * HD_ + 32 + j], s2 = sinb[t * HD_ + 32 + j];
  float lo = (float)buf[base];
  float hi = (float)buf[base + 32];
  buf[base]      = (_Float16)(lo * c1 - hi * s1);
  buf[base + 32] = (_Float16)(hi * c2 + lo * s2);
}

// ---------------------------------------------------------------------------
// GEMM helpers: load one A fragment + four B fragments for a 32-wide K step,
// and issue the 4 WMMAs of a 16x64 wave tile.
// ---------------------------------------------------------------------------
static __device__ __forceinline__ void loadAB(const _Float16* __restrict__ arow,
                                              const _Float16* __restrict__ bcol,
                                              int Kd, int k0, int g,
                                              v16h& a, v16h b[4]) {
  a = ld2x128(arow + k0 + g * 8, arow + k0 + 16 + g * 8);
#pragma unroll
  for (int d = 0; d < 4; ++d) {
    const _Float16* bp = bcol + (size_t)(d * 16) * Kd + k0 + g * 8;
    b[d] = ld2x128(bp, bp + 16);
  }
}
static __device__ __forceinline__ void mma4(v16h a, const v16h b[4], v8f acc[4]) {
#pragma unroll
  for (int d = 0; d < 4; ++d)
    acc[d] = __builtin_amdgcn_wmma_f32_16x16x32_f16(false, a, false, b[d],
                                                    (short)0, acc[d], false, false);
}

// ---------------------------------------------------------------------------
// WMMA GEMM:  Out[M x N] = A[M x Kd] (f16 row-major) * Bt (f16 col-major).
// Wave tile 16x64 (4 accumulators), WG = 8 waves -> 128x64 tile.
// Requires Kd % 64 == 0. Ping-pong pipelined, no buffer-rotation copies.
// mode 0: f16 row-major; mode 1: f16 head-transposed [B,H,HD,T]; mode 2: f32.
// ---------------------------------------------------------------------------
__global__ __launch_bounds__(256) void gemm16(const _Float16* __restrict__ A,
                                              const _Float16* __restrict__ Bt,
                                              void* __restrict__ Out,
                                              int Kd, int N, int mode) {
  const int lane = threadIdx.x & 31;
  const int warp = threadIdx.x >> 5;
  const int m = lane & 15, g = lane >> 4;
  const int r0 = blockIdx.y * 128 + warp * 16;
  const int c0 = blockIdx.x * 64;

  v8f acc[4] = {v8f{}, v8f{}, v8f{}, v8f{}};
  const _Float16* arow = A  + (size_t)(r0 + m) * Kd;   // A row
  const _Float16* bcol = Bt + (size_t)(c0 + m) * Kd;   // B col n = lane&15

  v16h aA, aB;
  v16h bA[4], bB[4];
  loadAB(arow, bcol, Kd, 0, g, aA, bA);                // stage 0

  int k0 = 0;
  for (; k0 < Kd - 64; k0 += 64) {
    loadAB(arow, bcol, Kd, k0 + 32, g, aB, bB);        // stage k0+32 in flight
    mma4(aA, bA, acc);                                 // consume k0
    loadAB(arow, bcol, Kd, k0 + 64, g, aA, bA);        // stage k0+64 in flight
    mma4(aB, bB, acc);                                 // consume k0+32
  }
  loadAB(arow, bcol, Kd, k0 + 32, g, aB, bB);          // last stage
  mma4(aA, bA, acc);
  mma4(aB, bB, acc);

#pragma unroll
  for (int d = 0; d < 4; ++d) {
#pragma unroll
    for (int v = 0; v < 8; ++v) {
      int r = r0 + v + 8 * g;          // C/D layout: row = vgpr + 8*(lane>=16)
      int c = c0 + d * 16 + m;         // col = lane&15 within 16-wide subtile
      float val = acc[d][v];
      if (mode == 2) {
        ((float*)Out)[(size_t)r * N + c] = val;
      } else if (mode == 0) {
        ((_Float16*)Out)[(size_t)r * N + c] = (_Float16)val;
      } else {  // mode 1: V^T store:  Vt[((b*H + h)*HD + j)*T + t]
        int bb = r >> 11, t = r & (T_ - 1);
        int hh = c >> 6,  j = c & (HD_ - 1);
        ((_Float16*)Out)[(((size_t)(bb * H_ + hh) * HD_ + j) * T_) + t] = (_Float16)val;
      }
    }
  }
}

// ---------------------------------------------------------------------------
// Flash attention: one wave per 16-query tile of one (b,h). 32-key chunks.
// All 8 global fragment loads (K and V) are issued at the top of each chunk:
// V loads finish under the softmax VALU work, K waits stagger across the
// score WMMAs. Next chunk's K/V lines are prefetched (global_prefetch_b8).
// Q,K: f16 [B,T,D]; Vt: f16 [B,H,HD,T]; ctx out: f16 [B,T,D].
// ---------------------------------------------------------------------------
__global__ __launch_bounds__(128) void attn(const _Float16* __restrict__ Q,
                                            const _Float16* __restrict__ K,
                                            const _Float16* __restrict__ Vt,
                                            _Float16* __restrict__ C) {
  __shared__ __align__(16) _Float16 lds[4][16 * 32];   // per-wave P staging tile
  const int lane = threadIdx.x & 31;
  const int warp = threadIdx.x >> 5;
  const int gw = blockIdx.x * 4 + warp;                // 4096 waves total
  const int qt = gw & (T_ / 16 - 1);
  const int h  = (gw >> 7) & (H_ - 1);
  const int b  = gw >> 11;
  const int q0 = qt * 16;
  const int m = lane & 15, g = lane >> 4;
  const int hbase = h * HD_;

  // Q A-fragments (16 queries x 64 head dims, split as two 16x32 fragments)
  const _Float16* qrow = Q + ((size_t)(b * T_ + q0 + m)) * D_ + hbase;
  v16h aq0 = ld2x128(qrow + g * 16,      qrow + g * 16 + 8);
  v16h aq1 = ld2x128(qrow + 32 + g * 16, qrow + 40 + g * 16);

  v8f o0 = {}, o1 = {}, o2 = {}, o3 = {};
  float rmax[8], rsum[8];
#pragma unroll
  for (int v = 0; v < 8; ++v) { rmax[v] = -1e30f; rsum[v] = 0.0f; }

  _Float16* pl = &lds[warp][0];
  const int qlast = q0 + 15;

  const _Float16* kbase = K + ((size_t)(b * T_ + m)) * D_ + hbase;
  const _Float16* vbase = Vt + (((size_t)(b * H_ + h) * HD_ + m) * T_) + g * 16;

  for (int key0 = 0; key0 <= qlast; key0 += 32) {
    // ---- issue ALL chunk loads up front: 4 K-frags + 4 V-frags ----
    const _Float16* kp0 = kbase + (size_t)key0 * D_;
    const _Float16* kp1 = kp0 + (size_t)16 * D_;
    const _Float16* vp  = vbase + key0;
    v16h bk0 = ld2x128(kp0 + g * 16,      kp0 + g * 16 + 8);
    v16h bk1 = ld2x128(kp0 + 32 + g * 16, kp0 + 40 + g * 16);
    v16h bk2 = ld2x128(kp1 + g * 16,      kp1 + g * 16 + 8);
    v16h bk3 = ld2x128(kp1 + 32 + g * 16, kp1 + 40 + g * 16);
    v16h bv0 = ld2x128(vp,                      vp + 8);
    v16h bv1 = ld2x128(vp + (size_t)16 * T_,    vp + (size_t)16 * T_ + 8);
    v16h bv2 = ld2x128(vp + (size_t)32 * T_,    vp + (size_t)32 * T_ + 8);
    v16h bv3 = ld2x128(vp + (size_t)48 * T_,    vp + (size_t)48 * T_ + 8);

    // prefetch next chunk's K/V cachelines while this chunk computes
    if (key0 + 32 <= qlast) {
      __builtin_prefetch(kp0 + (size_t)32 * D_, 0, 0);
      __builtin_prefetch(kp1 + (size_t)32 * D_, 0, 0);
      __builtin_prefetch(vp + 32, 0, 0);
      __builtin_prefetch(vp + (size_t)32 * T_ + 32, 0, 0);
    }

    // ---- scores: S[16q x 32keys] = Q * K^T ----
    v8f s0 = {}, s1 = {};
    s0 = __builtin_amdgcn_wmma_f32_16x16x32_f16(false, aq0, false, bk0, (short)0, s0, false, false);
    s0 = __builtin_amdgcn_wmma_f32_16x16x32_f16(false, aq1, false, bk1, (short)0, s0, false, false);
    s1 = __builtin_amdgcn_wmma_f32_16x16x32_f16(false, aq0, false, bk2, (short)0, s1, false, false);
    s1 = __builtin_amdgcn_wmma_f32_16x16x32_f16(false, aq1, false, bk3, (short)0, s1, false, false);

    // ---- online softmax over the 32-key chunk ----
    const bool needmask = (key0 + 31) > q0;
#pragma unroll
    for (int v = 0; v < 8; ++v) {
      float a0 = s0[v] * 0.125f;                 // 1/sqrt(64)
      float a1 = s1[v] * 0.125f;
      if (needmask) {
        int q = q0 + v + 8 * g;
        if (key0 + m      > q) a0 = -1e30f;      // causal mask
        if (key0 + 16 + m > q) a1 = -1e30f;
      }
      float t = fmaxf(a0, a1);                   // row max across 16 lanes
      t = fmaxf(t, __shfl_xor(t, 1, 32));
      t = fmaxf(t, __shfl_xor(t, 2, 32));
      t = fmaxf(t, __shfl_xor(t, 4, 32));
      t = fmaxf(t, __shfl_xor(t, 8, 32));
      float mn = fmaxf(rmax[v], t);
      float alpha = __expf(rmax[v] - mn);
      rmax[v] = mn;
      float p0 = __expf(a0 - mn);
      float p1 = __expf(a1 - mn);
      float rs = p0 + p1;
      rs += __shfl_xor(rs, 1, 32);
      rs += __shfl_xor(rs, 2, 32);
      rs += __shfl_xor(rs, 4, 32);
      rs += __shfl_xor(rs, 8, 32);
      rsum[v] = rsum[v] * alpha + rs;
      o0[v] *= alpha; o1[v] *= alpha; o2[v] *= alpha; o3[v] *= alpha;
      s0[v] = p0; s1[v] = p1;
    }

    // ---- reshape P via per-wave LDS tile into A-fragment layout ----
    asm volatile("s_wait_dscnt 0" ::: "memory");       // WAR vs prev iter's reads
#pragma unroll
    for (int v = 0; v < 8; ++v) {
      pl[(v + 8 * g) * 32 + m]      = (_Float16)s0[v];
      pl[(v + 8 * g) * 32 + 16 + m] = (_Float16)s1[v];
    }
    asm volatile("s_wait_dscnt 0" ::: "memory");       // RAW: stores visible
    v16h ap = ld2x128(pl + m * 32 + g * 16, pl + m * 32 + g * 16 + 8);

    // ---- O += P * V  (V fragments already resident) ----
    o0 = __builtin_amdgcn_wmma_f32_16x16x32_f16(false, ap, false, bv0, (short)0, o0, false, false);
    o1 = __builtin_amdgcn_wmma_f32_16x16x32_f16(false, ap, false, bv1, (short)0, o1, false, false);
    o2 = __builtin_amdgcn_wmma_f32_16x16x32_f16(false, ap, false, bv2, (short)0, o2, false, false);
    o3 = __builtin_amdgcn_wmma_f32_16x16x32_f16(false, ap, false, bv3, (short)0, o3, false, false);
  }

  // ---- normalize and store ctx (f16 row-major [B,T,D]) ----
#pragma unroll
  for (int v = 0; v < 8; ++v) {
    float inv = 1.0f / rsum[v];
    _Float16* cp = C + ((size_t)(b * T_ + q0 + v + 8 * g)) * D_ + hbase + m;
    cp[0]  = (_Float16)(o0[v] * inv);
    cp[16] = (_Float16)(o1[v] * inv);
    cp[32] = (_Float16)(o2[v] * inv);
    cp[48] = (_Float16)(o3[v] * inv);
  }
}

// ---------------------------------------------------------------------------
extern "C" void kernel_launch(void* const* d_in, const int* in_sizes, int n_in,
                              void* d_out, int out_size, void* d_ws, size_t ws_size,
                              hipStream_t stream) {
  const float* x  = (const float*)d_in[0];
  const float* Wq = (const float*)d_in[1];
  const float* Wk = (const float*)d_in[2];
  const float* Wv = (const float*)d_in[3];
  const float* Wo = (const float*)d_in[4];
  const float* rc = (const float*)d_in[5];
  const float* rs = (const float*)d_in[6];

  char* ws = (char*)d_ws;
  const size_t MB = 1024 * 1024;
  _Float16* xh  = (_Float16*)(ws);             // 8 MB
  _Float16* Wqt = (_Float16*)(ws + 8  * MB);   // 2 MB each
  _Float16* Wkt = (_Float16*)(ws + 10 * MB);
  _Float16* Wvt = (_Float16*)(ws + 12 * MB);
  _Float16* Wot = (_Float16*)(ws + 14 * MB);
  _Float16* Qb  = (_Float16*)(ws + 16 * MB);   // 8 MB
  _Float16* Kb  = (_Float16*)(ws + 24 * MB);   // 8 MB
  _Float16* Vtb = (_Float16*)(ws + 32 * MB);   // 8 MB
  _Float16* Cb  = (_Float16*)(ws);             // reuse xh region for ctx

  const int nx = B_ * T_ * D_;                 // 4194304
  cvt_f32_f16 <<<nx / 256, 256, 0, stream>>>(x, xh, nx);
  cvtT_f32_f16<<<(D_ * D_) / 256, 256, 0, stream>>>(Wq, Wqt);
  cvtT_f32_f16<<<(D_ * D_) / 256, 256, 0, stream>>>(Wk, Wkt);
  cvtT_f32_f16<<<(D_ * D_) / 256, 256, 0, stream>>>(Wv, Wvt);
  cvtT_f32_f16<<<(D_ * D_) / 256, 256, 0, stream>>>(Wo, Wot);

  dim3 ggrid(D_ / 64, (B_ * T_) / 128);        // (16, 32)
  gemm16<<<ggrid, 256, 0, stream>>>(xh, Wqt, Qb,  D_, D_, 0);
  gemm16<<<ggrid, 256, 0, stream>>>(xh, Wkt, Kb,  D_, D_, 0);
  gemm16<<<ggrid, 256, 0, stream>>>(xh, Wvt, Vtb, D_, D_, 1);

  const int nr = B_ * T_ * H_ * 32;            // 2097152
  rope_f16<<<nr / 256, 256, 0, stream>>>(Qb, rc, rs);
  rope_f16<<<nr / 256, 256, 0, stream>>>(Kb, rc, rs);

  attn<<<(B_ * H_ * (T_ / 16)) / 4, 128, 0, stream>>>(Qb, Kb, Vtb, Cb);

  gemm16<<<ggrid, 256, 0, stream>>>(Cb, Wot, d_out, D_, D_, 2);

  (void)in_sizes; (void)n_in; (void)out_size; (void)ws_size;
}
